// GraphormerAttention_26542897889977
// MI455X (gfx1250) — matrix operand order repaired and checked by reading, not compiled, not run on hardware
//
#include <hip/hip_runtime.h>
#include <hip/hip_bf16.h>

// GraphormerAttention on MI455X (gfx1250, wave32, WMMA).
// Roofline: ~10.7 GFLOP vs ~30 MB unique HBM traffic -> matrix-op bound.
// Path: f16 WMMA 16x16x32 with f32 accumulation for all four GEMMs + both
// attention einsums; softmax computed without max-subtraction (shift-invariant,
// logits ~N(0,1) so exp() cannot overflow in f32). z-bins precomputed once as
// u8 so the 8 heads don't each re-read the 16MB f32 z_matrix.

#define D_MODEL   512
#define NUM_HEADS 8
#define HEAD_DIM  64
#define N_TOK     2048
#define NUM_Z_BINS 16

typedef _Float16 h8   __attribute__((ext_vector_type(8)));
typedef _Float16 v16h __attribute__((ext_vector_type(16)));
typedef float    v8f  __attribute__((ext_vector_type(8)));
typedef float    v4f  __attribute__((ext_vector_type(4)));

union Frag16 { v16h v; h8 h[2]; };

// A-matrix frag (16x32 f16, row-major source). Per CDNA5 ISA 16-bit A layout:
// lanes 0-15 hold rows 0-15 with K = {0..7, 16..23}; lanes 16-31 same rows with
// K = {8..15, 24..31}. Element j -> k = (j>=8 ? 16 : 0) + half*8 + (j&7),
// i.e. two contiguous 16-byte chunks per lane.
static __device__ __forceinline__ v16h load_a_frag(const _Float16* __restrict__ A,
                                                   int lda, int row0, int kb, int lane) {
  const int r = lane & 15, hf = lane >> 4;
  const _Float16* p = A + (size_t)(row0 + r) * lda + kb + hf * 8;
  Frag16 f;
  f.h[0] = *(const h8*)(p);
  f.h[1] = *(const h8*)(p + 16);
  return f.v;
}

// B-matrix frag (32x16 f16). Source Bt is column-major of the logical B, i.e.
// Bt[n][k]. Same K order as the A frag: lanes 0-15 hold cols 0-15 with
// K = {0..7, 16..23}; lanes 16-31 hold K = {8..15, 24..31}.
// Two contiguous 16-byte chunks per lane.
static __device__ __forceinline__ v16h load_b_frag(const _Float16* __restrict__ Bt,
                                                   int ldb, int col0, int kb, int lane) {
  const int c = lane & 15, hf = lane >> 4;
  const _Float16* p = Bt + (size_t)(col0 + c) * ldb + kb + hf * 8;
  Frag16 f;
  f.h[0] = *(const h8*)(p);
  f.h[1] = *(const h8*)(p + 16);
  return f.v;
}

__global__ void cvt_f16_kernel(const float* __restrict__ in, _Float16* __restrict__ out, int n) {
  int i = blockIdx.x * blockDim.x + threadIdx.x;
  if (i < n) out[i] = (_Float16)in[i];
}

// 512x512 f32 row-major -> f16 transposed (Bt[n][k] = W[k][n])
__global__ void transpose_f16_kernel(const float* __restrict__ in, _Float16* __restrict__ out) {
  int i = blockIdx.x * blockDim.x + threadIdx.x;
  int r = i >> 9, c = i & 511;
  out[(size_t)c * D_MODEL + r] = (_Float16)in[(size_t)r * D_MODEL + c];
}

// z_matrix (f32) -> u8 bin indices, head-independent, computed once.
// 4 elements per thread (vectorized b128 load, b32 store).
__global__ void zbin_kernel(const float* __restrict__ z, unsigned char* __restrict__ zb, int n4) {
  int i = blockIdx.x * blockDim.x + threadIdx.x;
  if (i >= n4) return;
  const v4f zv = *(const v4f*)(z + (size_t)i * 4);
  const float binscale = (float)NUM_Z_BINS / 5.0f;   // NUM_Z_BINS / MAX_Z
  unsigned int packed = 0;
#pragma unroll
  for (int j = 0; j < 4; ++j) {
    int b = (int)floorf(zv[j] * binscale);
    b = b < 0 ? 0 : (b > 15 ? 15 : b);
    packed |= ((unsigned int)b) << (8 * j);
  }
  *(unsigned int*)(zb + (size_t)i * 4) = packed;
}

// GEMM: [2048 x 512] (f16 row-major A) x Bt (f16, [512 out][512 in]) + bias.
// One wave computes a 16x64 output tile: 16 k-steps x 4 v_wmma_f32_16x16x32_f16.
// MODE 0: f16 row-major out; MODE 1: f16 transposed out (for V: [d_global][m]);
// MODE 2: f32 row-major out (final projection into d_out).
template <int MODE>
__global__ __launch_bounds__(256) void wmma_gemm_16x64(
    const _Float16* __restrict__ A, const _Float16* __restrict__ Bt,
    const float* __restrict__ bias, _Float16* __restrict__ outH,
    float* __restrict__ outF)
{
  const int wid  = (int)((blockIdx.x * blockDim.x + threadIdx.x) >> 5);
  const int lane = threadIdx.x & 31;
  const int cl = lane & 15, hf = lane >> 4;
  const int n0 = (wid >> 3) * 16;   // 128 row tiles
  const int c0 = (wid & 7) * 64;    // 8 col tiles of 64

  v8f acc[4] = {};
  for (int kb = 0; kb < D_MODEL; kb += 32) {
    const v16h a = load_a_frag(A, D_MODEL, n0, kb, lane);
#pragma unroll
    for (int t = 0; t < 4; ++t) {
      const v16h b = load_b_frag(Bt, D_MODEL, c0 + t * 16, kb, lane);
      acc[t] = __builtin_amdgcn_wmma_f32_16x16x32_f16(false, a, false, b,
                                                      (short)0, acc[t], false, false);
    }
  }
  // C/D layout: VGPR g, lane -> row n0 + g + 8*(lane>=16), col c0 + t*16 + (lane&15)
#pragma unroll
  for (int t = 0; t < 4; ++t) {
    const int c = c0 + t * 16 + cl;
    const float bv = bias[c];
#pragma unroll
    for (int g = 0; g < 8; ++g) {
      const int n = n0 + g + hf * 8;
      const float v = acc[t][g] + bv;
      if (MODE == 0)      outH[(size_t)n * D_MODEL + c] = (_Float16)v;
      else if (MODE == 1) outH[(size_t)c * N_TOK + n]   = (_Float16)v;  // Vt[c][n]
      else                outF[(size_t)n * D_MODEL + c] = v;
    }
  }
}

// Attention: one wave owns a 16-query tile for one head, streams keys in
// m-tiles of 32. Per m-tile: 4 WMMAs (S = Q K^T), u8 z-bin load + LDS table
// bias, exp, P bounced through 1KB per-wave LDS (C-layout -> A-layout),
// 4 WMMAs (acc += P V). V-frags are loaded before the exp/LDS section so the
// global loads fly under the VALU work and the ds fence.
__global__ __launch_bounds__(256) void attn_kernel(
    const _Float16* __restrict__ Q, const _Float16* __restrict__ Kmat,
    const _Float16* __restrict__ Vt, const unsigned char* __restrict__ zb,
    const float* __restrict__ ztab, _Float16* __restrict__ outA)
{
  __shared__ float zt[NUM_Z_BINS];
  __shared__ _Float16 ps_all[8][16 * 32];   // per-wave P scratch (16 rows x 32 m)

  const int h    = blockIdx.y;
  const int wv   = threadIdx.x >> 5;
  const int lane = threadIdx.x & 31;
  const int cl = lane & 15, hf = lane >> 4;
  const int n0 = (blockIdx.x * 8 + wv) * 16;

  if (threadIdx.x < NUM_Z_BINS) zt[threadIdx.x] = ztab[threadIdx.x * NUM_HEADS + h];
  __syncthreads();

  const _Float16* Qh = Q    + h * HEAD_DIM;              // row stride D_MODEL
  const _Float16* Kb = Kmat + h * HEAD_DIM;              // K row-major == Bt for S
  const _Float16* Vh = Vt   + (size_t)(h * HEAD_DIM) * N_TOK;  // Vt[d][m]

  // Persistent Q A-frags covering d = 0..63 (two k-blocks of 32)
  const v16h qa0 = load_a_frag(Qh, D_MODEL, n0, 0, lane);
  const v16h qa1 = load_a_frag(Qh, D_MODEL, n0, 32, lane);

  v8f acc[4] = {};
  float rs[8] = {0.f, 0.f, 0.f, 0.f, 0.f, 0.f, 0.f, 0.f};
  _Float16* ps = &ps_all[wv][0];
  const float scale = 0.125f;                     // HEAD_DIM^-0.5

  for (int m0 = 0; m0 < N_TOK; m0 += 32) {
    // ---- S = Q K^T for 16x32 tile (two 16x16 C-tiles, K=64) ----
    v8f s0 = {}, s1 = {};
    v16h b;
    b = load_b_frag(Kb, D_MODEL, m0,      0,  lane);
    s0 = __builtin_amdgcn_wmma_f32_16x16x32_f16(false, qa0, false, b, (short)0, s0, false, false);
    b = load_b_frag(Kb, D_MODEL, m0,      32, lane);
    s0 = __builtin_amdgcn_wmma_f32_16x16x32_f16(false, qa1, false, b, (short)0, s0, false, false);
    b = load_b_frag(Kb, D_MODEL, m0 + 16, 0,  lane);
    s1 = __builtin_amdgcn_wmma_f32_16x16x32_f16(false, qa0, false, b, (short)0, s1, false, false);
    b = load_b_frag(Kb, D_MODEL, m0 + 16, 32, lane);
    s1 = __builtin_amdgcn_wmma_f32_16x16x32_f16(false, qa1, false, b, (short)0, s1, false, false);

    // ---- hoist V-tile B-frags (independent of S / P staging) ----
    v16h vf[4];
#pragma unroll
    for (int t = 0; t < 4; ++t) {
      Frag16 f;
      const _Float16* p = Vh + (size_t)(t * 16 + cl) * N_TOK + m0 + hf * 8;
      f.h[0] = *(const h8*)p;
      f.h[1] = *(const h8*)(p + 16);
      vf[t] = f.v;
    }

    // ---- z-bin bias, exp, rowsum, stage P (f16) into LDS ----
#pragma unroll
    for (int g = 0; g < 8; ++g) {
      const int n = n0 + g + hf * 8;
      const unsigned char* zrow = zb + (size_t)n * N_TOK + m0;
      const int i0 = zrow[cl];            // precomputed bins, coalesced b8 reads
      const int i1 = zrow[cl + 16];
      const float p0 = __expf(s0[g] * scale + zt[i0]);
      const float p1 = __expf(s1[g] * scale + zt[i1]);
      rs[g] += p0 + p1;
      const int row = g + hf * 8;
      ps[row * 32 + cl]      = (_Float16)p0;
      ps[row * 32 + cl + 16] = (_Float16)p1;
    }
    __threadfence_block();  // LDS in-order per wave; fence blocks compiler reordering

    // ---- P as A-frag (16x32 f16) from LDS, then acc += P * V-tile ----
    v16h pa;
    {
      Frag16 f;
      const _Float16* pp = ps + cl * 32 + hf * 8;
      f.h[0] = *(const h8*)(pp);
      f.h[1] = *(const h8*)(pp + 16);
      pa = f.v;
    }
#pragma unroll
    for (int t = 0; t < 4; ++t) {
      acc[t] = __builtin_amdgcn_wmma_f32_16x16x32_f16(false, pa, false, vf[t],
                                                      (short)0, acc[t], false, false);
    }
  }

  // ---- row-sum reduction across the 16 lanes of each half-group ----
  // Row mapping of rs[g] (n0+g+8*hf) matches the C-layout of acc exactly.
#pragma unroll
  for (int g = 0; g < 8; ++g) {
    float s = rs[g];
    s += __shfl_xor(s, 1, 32);
    s += __shfl_xor(s, 2, 32);
    s += __shfl_xor(s, 4, 32);
    s += __shfl_xor(s, 8, 32);
    rs[g] = s;
  }

  // ---- normalize and store head output (f16, row-major [n][D_MODEL]) ----
#pragma unroll
  for (int t = 0; t < 4; ++t) {
#pragma unroll
    for (int g = 0; g < 8; ++g) {
      const int n = n0 + g + hf * 8;
      const int c = h * HEAD_DIM + t * 16 + cl;
      outA[(size_t)n * D_MODEL + c] = (_Float16)(acc[t][g] / rs[g]);
    }
  }
}

extern "C" void kernel_launch(void* const* d_in, const int* in_sizes, int n_in,
                              void* d_out, int out_size, void* d_ws, size_t ws_size,
                              hipStream_t stream) {
  const float* x    = (const float*)d_in[0];
  const float* zmat = (const float*)d_in[1];
  const float* Wq = (const float*)d_in[2];
  const float* bq = (const float*)d_in[3];
  const float* Wk = (const float*)d_in[4];
  const float* bk = (const float*)d_in[5];
  const float* Wv = (const float*)d_in[6];
  const float* bv = (const float*)d_in[7];
  const float* Wo = (const float*)d_in[8];
  const float* bo = (const float*)d_in[9];
  const float* zt = (const float*)d_in[10];

  char* ws = (char*)d_ws;
  const size_t MB = (size_t)1 << 20;
  _Float16* xh  = (_Float16*)(ws);                       // 2 MB: x in f16
  _Float16* WqT = (_Float16*)(ws + 2 * MB);              // 512 KB each, transposed f16
  _Float16* WkT = (_Float16*)(ws + 2 * MB + 512 * 1024);
  _Float16* WvT = (_Float16*)(ws + 3 * MB);
  _Float16* WoT = (_Float16*)(ws + 3 * MB + 512 * 1024);
  _Float16* Qm  = (_Float16*)(ws + 4 * MB);              // 2 MB [n][512]
  _Float16* Km  = (_Float16*)(ws + 6 * MB);              // 2 MB [m][512]
  _Float16* Vt  = (_Float16*)(ws + 8 * MB);              // 2 MB [512][m] (transposed)
  _Float16* Ao  = (_Float16*)(ws + 10 * MB);             // 2 MB attention output f16
  unsigned char* zbins = (unsigned char*)(ws + 12 * MB); // 4 MB u8 bins

  // Pack inputs to f16 / transposed-f16 / u8 bins
  cvt_f16_kernel<<<(N_TOK * D_MODEL) / 256, 256, 0, stream>>>(x, xh, N_TOK * D_MODEL);
  transpose_f16_kernel<<<(D_MODEL * D_MODEL) / 256, 256, 0, stream>>>(Wq, WqT);
  transpose_f16_kernel<<<(D_MODEL * D_MODEL) / 256, 256, 0, stream>>>(Wk, WkT);
  transpose_f16_kernel<<<(D_MODEL * D_MODEL) / 256, 256, 0, stream>>>(Wv, WvT);
  transpose_f16_kernel<<<(D_MODEL * D_MODEL) / 256, 256, 0, stream>>>(Wo, WoT);
  zbin_kernel<<<(N_TOK * N_TOK / 4) / 256, 256, 0, stream>>>(zmat, zbins, N_TOK * N_TOK / 4);

  // Q/K/V projections (1024 waves each, 16x64 tiles)
  wmma_gemm_16x64<0><<<128, 256, 0, stream>>>(xh, WqT, bq, Qm, nullptr);
  wmma_gemm_16x64<0><<<128, 256, 0, stream>>>(xh, WkT, bk, Km, nullptr);
  wmma_gemm_16x64<1><<<128, 256, 0, stream>>>(xh, WvT, bv, Vt, nullptr);

  // Attention: 128 query tiles x 8 heads = 1024 waves
  attn_kernel<<<dim3(16, 8), 256, 0, stream>>>(Qm, Km, Vt, zbins, zt, Ao);

  // Output projection -> f32 d_out
  wmma_gemm_16x64<2><<<128, 256, 0, stream>>>(Ao, WoT, bo, nullptr, (float*)d_out);
}
